// _ConvLSTM_26096221291123
// MI455X (gfx1250) — matrix operand order repaired and checked by reading, not compiled, not run on hardware
//
#include <hip/hip_runtime.h>

typedef __attribute__((ext_vector_type(16))) _Float16 v16h;
typedef __attribute__((ext_vector_type(8)))  float    v8f;

#define HID    64
#define KSTEPS 36          // K = 9 taps * 128 ch = 1152 = 36 * 32  (tap-major ordering)
#define NPX    64          // pixels per block tile = one full row
#define CSTR   144         // padded per-pixel channel stride in LDS (f16 units, 288 B)
#define SIN_ELEMS (3 * 66 * CSTR)            // 28,512 f16 = 57,024 B
#define SMEM_BYTES 65536                     // overlay: max(57,024, 256*64*4)

__global__ void zero_u32_kernel(unsigned* __restrict__ p, size_t n) {
  size_t i = (size_t)blockIdx.x * blockDim.x + threadIdx.x;
  if (i < n) p[i] = 0u;
}

// Re-lay W [256][128][3][3] f32 into exact WMMA A-fragment register image (f16),
// with tap-major K ordering: k = (r*3+s)*128 + ic.
// Apack[kb][mt][lane][e];  e -> (vgpr v = e/2, half = e%2); lane-group g = lane>>4.
__global__ void pack_weights_kernel(const float* __restrict__ W,
                                    _Float16* __restrict__ Apack) {
  int kb = blockIdx.x >> 4;        // 0..35
  int mt = blockIdx.x & 15;        // 0..15
  int lane = threadIdx.x;          // 0..31
  int m = mt * 16 + (lane & 15);   // output channel 0..255
  int g = lane >> 4;
  _Float16* dst = Apack + ((((size_t)kb * 16 + mt) * 32) + lane) * 16;
#pragma unroll
  for (int e = 0; e < 16; ++e) {
    int v = e >> 1, hh = e & 1;
    int klocal = g * 16 + 2 * v + hh;
    int k   = kb * 32 + klocal;    // 0..1151 ; k = tap*128 + ic
    int tap = k >> 7;              // 0..8
    int ic  = k & 127;
    int r   = tap / 3;
    int s   = tap - r * 3;
    dst[e] = (_Float16)W[(((size_t)m * 128 + ic) * 3 + r) * 3 + s];
  }
}

// One ConvLSTM cell step: 256(M) x 64(N) tile (one image row) + cell update.
// Input channels 0..63 from x (f32 layer0 / f16 layer1), 64..127 from h_prev (f16).
__global__ __launch_bounds__(256) void convlstm_step_kernel(
    const float*    __restrict__ x_f32,   // layer0 input or nullptr
    const _Float16* __restrict__ x_f16,   // layer1 input or nullptr
    const _Float16* __restrict__ h_prev,  // [4,64,64,64] f16
    float*          __restrict__ c,       // [4,64,64,64] f32, in place
    const _Float16* __restrict__ Apack,   // packed weights
    const float*    __restrict__ bias,    // [256] f32
    _Float16*       __restrict__ h_next,  // [4,64,64,64] f16
    float*          __restrict__ out_f32) // nullptr for layer0
{
  // Dynamic LDS, overlaid: halo patch during the K loop, gate tile afterwards.
  extern __shared__ __align__(32) char smem[];
  _Float16* sIn  = (_Float16*)smem;   // [3 rows][66 x][CSTR ch], 57,024 B
  float*   sFeat = (float*)smem;      // [256 oc][64 px], 65,536 B (after barrier)

  const int tid = threadIdx.x;
  const int blk = blockIdx.x;                 // 0..255
  const int b   = blk >> 6;                   // batch
  const int y   = blk & 63;                   // row

  // ---- stage halo patch (3 rows x 66 px x 128 ch) into LDS, f16 ----
  for (int idx = tid; idx < 128 * 3 * 66; idx += 256) {
    int j   = idx % 66;                       // consecutive tids -> consecutive x: coalesced
    int tmp = idx / 66;
    int r   = tmp % 3;
    int ic  = tmp / 3;
    int yg  = y + r - 1;
    int xg  = j - 1;
    _Float16 val = (_Float16)0.0f;
    if ((unsigned)yg < 64u && (unsigned)xg < 64u) {
      size_t off = ((((size_t)b * 64 + (ic & 63)) * 64 + yg) * 64) + xg;
      if (ic < 64) {
        val = x_f32 ? (_Float16)x_f32[off] : x_f16[off];
      } else {
        val = h_prev[off];
      }
    }
    sIn[(r * 66 + j) * CSTR + ic] = val;
  }
  __syncthreads();

  const int lane = tid & 31;
  const int wave = tid >> 5;                  // 0..7 -> M-tiles {2w, 2w+1}
  const int lg   = lane >> 4;                 // lane group
  const int ln   = lane & 15;

  v8f acc[2][4];
#pragma unroll
  for (int i = 0; i < 2; ++i)
#pragma unroll
    for (int n = 0; n < 4; ++n) acc[i][n] = (v8f){};

  // ---- K loop: 9 taps x 4 channel chunks of 32; 8 WMMAs (2M x 4N) per chunk ----
  for (int tap = 0; tap < 9; ++tap) {
    const int r = tap / 3;                    // uniform (SALU)
    const int s = tap - r * 3;
    const _Float16* rowb = &sIn[(r * 66 + ln + s) * CSTR + lg * 16];
#pragma unroll
    for (int q = 0; q < 4; ++q) {
      const int kb = tap * 4 + q;
      // A fragments: one 32B coalesced load per lane (pre-packed register image)
      const v16h fa0 = *(const v16h*)(Apack + ((((size_t)kb * 16 + 2 * wave)     * 32) + lane) * 16);
      const v16h fa1 = *(const v16h*)(Apack + ((((size_t)kb * 16 + 2 * wave + 1) * 32) + lane) * 16);
      if (kb + 1 < KSTEPS)
        __builtin_prefetch(Apack + ((((size_t)(kb + 1) * 16 + 2 * wave) * 32) + lane) * 16, 0, 0);
      // B fragments: 16 consecutive channels -> 32B aligned LDS vector load, 4 N-tiles
      v16h fb[4];
#pragma unroll
      for (int n = 0; n < 4; ++n)
        fb[n] = *(const v16h*)(rowb + (n * 16) * CSTR + q * 32);

#pragma unroll
      for (int n = 0; n < 4; ++n) {
        acc[0][n] = __builtin_amdgcn_wmma_f32_16x16x32_f16(false, fa0, false, fb[n], (short)0, acc[0][n], false, false);
        acc[1][n] = __builtin_amdgcn_wmma_f32_16x16x32_f16(false, fa1, false, fb[n], (short)0, acc[1][n], false, false);
      }
    }
  }

  __syncthreads();   // halo patch dead; reuse LDS for the gate tile

  // ---- spill gate tile to LDS (C/D layout: VGPR rr -> M = rr + 8*lg, col = ln) ----
#pragma unroll
  for (int i = 0; i < 2; ++i) {
#pragma unroll
    for (int rr = 0; rr < 8; ++rr) {
      int m = (2 * wave + i) * 16 + rr + 8 * lg;
#pragma unroll
      for (int n = 0; n < 4; ++n)
        sFeat[m * NPX + n * 16 + ln] = acc[i][n][rr];
    }
  }
  __syncthreads();

  // ---- LSTM cell: gates i,f,o,g at oc = {ch, 64+ch, 128+ch, 192+ch} ----
  for (int idx = tid; idx < HID * NPX; idx += 256) {
    int ch = idx >> 6;
    int p  = idx & 63;
    float vi = sFeat[ch * NPX + p]         + bias[ch];
    float vf = sFeat[(64  + ch) * NPX + p] + bias[64  + ch];
    float vo = sFeat[(128 + ch) * NPX + p] + bias[128 + ch];
    float vg = sFeat[(192 + ch) * NPX + p] + bias[192 + ch];
    float gi = 1.0f / (1.0f + __expf(-vi));
    float gf = 1.0f / (1.0f + __expf(-vf));
    float go = 1.0f / (1.0f + __expf(-vo));
    float gg = tanhf(vg);
    size_t off = ((((size_t)b * 64 + ch) * 64 + y) * 64) + p;
    float cn = gf * c[off] + gi * gg;
    c[off] = cn;
    float hv = go * tanhf(cn);
    h_next[off] = (_Float16)hv;
    if (out_f32) out_f32[off] = hv;
  }
}

extern "C" void kernel_launch(void* const* d_in, const int* in_sizes, int n_in,
                              void* d_out, int out_size, void* d_ws, size_t ws_size,
                              hipStream_t stream) {
  const float* x  = (const float*)d_in[0];   // [10,4,64,64,64]
  const float* W0 = (const float*)d_in[1];   // [256,128,3,3]
  const float* B0 = (const float*)d_in[2];   // [256]
  const float* W1 = (const float*)d_in[3];
  const float* B1 = (const float*)d_in[4];
  float* out = (float*)d_out;                // [10,4,64,64,64]

  // workspace layout (bytes)
  char* ws = (char*)d_ws;
  float*    c0  = (float*)(ws + 0);           // 4,194,304
  float*    c1  = (float*)(ws + 4194304);     // 4,194,304
  _Float16* h0a = (_Float16*)(ws + 8388608);  // 2,097,152
  _Float16* h1a = (_Float16*)(ws + 10485760); // 2,097,152
  _Float16* h0b = (_Float16*)(ws + 12582912); // 2,097,152
  _Float16* h1b = (_Float16*)(ws + 14680064); // 2,097,152
  _Float16* A0  = (_Float16*)(ws + 16777216); // 589,824
  _Float16* A1  = (_Float16*)(ws + 16777216 + 589824);

  // zero c0, c1, h0a, h1a (contiguous first 12,582,912 B) — fresh state every call
  size_t nz = 12582912u / 4u;
  zero_u32_kernel<<<(unsigned)((nz + 255) / 256), 256, 0, stream>>>((unsigned*)ws, nz);

  pack_weights_kernel<<<KSTEPS * 16, 32, 0, stream>>>(W0, A0);
  pack_weights_kernel<<<KSTEPS * 16, 32, 0, stream>>>(W1, A1);

  _Float16 *h0p = h0a, *h0q = h0b, *h1p = h1a, *h1q = h1b;
  const size_t xs = (size_t)4 * 64 * 64 * 64;  // per-timestep stride (elements)

  for (int t = 0; t < 10; ++t) {
    convlstm_step_kernel<<<256, 256, SMEM_BYTES, stream>>>(
        x + (size_t)t * xs, nullptr, h0p, c0, A0, B0, h0q, nullptr);
    convlstm_step_kernel<<<256, 256, SMEM_BYTES, stream>>>(
        nullptr, h0q, h1p, c1, A1, B1, h1q, out + (size_t)t * xs);
    _Float16* tmp;
    tmp = h0p; h0p = h0q; h0q = tmp;
    tmp = h1p; h1p = h1q; h1q = tmp;
  }
}
